// GWNN_68341519613982
// MI455X (gfx1250) — matrix operand order repaired and checked by reading, not compiled, not run on hardware
//
#include <hip/hip_runtime.h>
#include <hip/hip_bf16.h>
#include <stdint.h>

// ---------------------------------------------------------------------------
// GWNN on MI455X (gfx1250, wave32).
// fp32 operands streamed from HBM (~1.03 GB of mandatory wavelet-matrix
// traffic -> ~45us floor @ 23.3 TB/s), converted to bf16 with the native
// packed converters (v_cvt_pk_bf16_f32), staged in LDS, and multiplied with
// v_wmma_f32_16x16x32_bf16 (f32 accumulation). Diag-filter scaling / ReLU
// fused into GEMM epilogues; log_softmax is a wave-per-row shuffle reduction.
// ---------------------------------------------------------------------------

typedef __attribute__((ext_vector_type(16))) __bf16 v16bf;
typedef __attribute__((ext_vector_type(8)))  __bf16 v8bf;
typedef __attribute__((ext_vector_type(4)))  __bf16 v4bf;
typedef __attribute__((ext_vector_type(2)))  __bf16 v2bf;
typedef __attribute__((ext_vector_type(8)))  float  v8f;
typedef __attribute__((ext_vector_type(4)))  float  v4f;
typedef __attribute__((ext_vector_type(2)))  float  v2f;

static constexpr int KT  = 32;        // bf16 WMMA K per step
static constexpr int LDK = KT + 8;    // LDS row pitch in bf16 elems (80B, 16B-aligned, bank-spread)

enum Epi { EPI_NONE = 0, EPI_ROWSCALE = 1, EPI_RELU = 2 };

// C[M,N] = epi( A[M,K] @ B[K,N] ), A/B/C fp32 in global, bf16 math in WMMA.
// Block: 256 threads = 8 waves, wave grid (BM/WM) x (BN/WN) == 8.
template <int BM, int BN, int WM, int WN, int EPI>
__global__ __launch_bounds__(256) void gwnn_gemm(
    const float* __restrict__ A, const float* __restrict__ B,
    float* __restrict__ C, const float* __restrict__ scale,
    int M, int N, int K) {
  constexpr int MFRAG = WM / 16;
  constexpr int NFRAG = WN / 16;
  constexpr int WROWS = BM / WM;
  constexpr int WCOLS = BN / WN;
  static_assert(WROWS * WCOLS == 8, "8 waves per block");
  constexpr int AREG = (BM * KT) / (4 * 256);   // A float4 chunks per thread
  constexpr int BREG = (BN * KT) / (4 * 256);   // B k-rows per thread (4 cols each)
  constexpr int BNG  = BN / 4;                  // n-groups of 4 columns

  __shared__ __bf16 ldsA[2][BM * LDK];  // [m][k]
  __shared__ __bf16 ldsB[2][BN * LDK];  // [n][k] (B transposed)

  const int tid    = threadIdx.x;
  const int lane   = tid & 31;
  const int waveid = tid >> 5;
  const int wr     = waveid % WROWS;
  const int wc     = waveid / WROWS;
  const int bm0    = blockIdx.y * BM;
  const int bn0    = blockIdx.x * BN;
  const int lrow   = lane & 15;   // M (A) or N (B) index within a fragment
  const int lhi    = lane >> 4;   // K-half selector per WMMA operand layout

  // B sub-panel owned by this thread: 4 consecutive columns, BREG consecutive
  // k-rows. Keeps global reads coalesced AND makes the [n][k] transpose a
  // packed-k store (vector cvt + ds_store_b64/b32, no half-reg shuffles).
  const int bn4 = tid % BNG;            // column group (4 cols)
  const int bk0 = (tid / BNG) * BREG;   // first k-row of this thread's panel

  v8f acc[MFRAG][NFRAG] = {};
  v4f aReg[AREG];
  v4f bReg[BREG];

  auto gloadA = [&](int k0) {
#pragma unroll
    for (int i = 0; i < AREG; ++i) {
      int c = tid + i * 256;
      int row = c >> 3;            // KT/4 = 8 float4 per row
      int kc  = c & 7;
      aReg[i] = *(const v4f*)(A + (size_t)(bm0 + row) * K + k0 + kc * 4);
    }
  };
  auto gloadB = [&](int k0) {
#pragma unroll
    for (int i = 0; i < BREG; ++i)
      bReg[i] = *(const v4f*)(B + (size_t)(k0 + bk0 + i) * N + bn0 + bn4 * 4);
  };
  // fp32 -> bf16 via native packed converters (vector fptrunc, RNE).
  auto stashA = [&](int buf) {
#pragma unroll
    for (int i = 0; i < AREG; ++i) {
      int c = tid + i * 256;
      int row = c >> 3;
      int kc  = c & 7;
      v4bf cv = __builtin_convertvector(aReg[i], v4bf);
      *(v4bf*)&ldsA[buf][row * LDK + kc * 4] = cv;   // 2x cvt_pk + ds_store_b64
    }
  };
  auto stashB = [&](int buf) {
    __bf16* p = &ldsB[buf][0];
#pragma unroll
    for (int j = 0; j < 4; ++j) {                    // for each owned column n
      const int n = bn4 * 4 + j;
      if constexpr (BREG % 4 == 0) {
#pragma unroll
        for (int i = 0; i < BREG; i += 4) {          // pack 4 adjacent k
          v4f t = {bReg[i][j], bReg[i + 1][j], bReg[i + 2][j], bReg[i + 3][j]};
          *(v4bf*)&p[n * LDK + bk0 + i] = __builtin_convertvector(t, v4bf);
        }
      } else {
#pragma unroll
        for (int i = 0; i < BREG; i += 2) {          // pack 2 adjacent k
          v2f t = {bReg[i][j], bReg[i + 1][j]};
          *(v2bf*)&p[n * LDK + bk0 + i] = __builtin_convertvector(t, v2bf);
        }
      }
    }
  };

  const int nk = K / KT;
  gloadA(0);
  gloadB(0);
  stashA(0);
  stashB(0);
  __syncthreads();

  for (int kt = 0; kt < nk; ++kt) {
    const int cur = kt & 1, nxt = cur ^ 1;
    if (kt + 1 < nk) {             // issue next tile's global loads early
      gloadA((kt + 1) * KT);
      gloadB((kt + 1) * KT);
    }

    const __bf16* As = &ldsA[cur][0];
    const __bf16* Bs = &ldsB[cur][0];

    // A fragment (16x32 bf16): lane L holds row M=L%16; K 0..7,16..23 (L<16)
    // or K 8..15,24..31 (L>=16) -> two contiguous b128 reads.
    v16bf afrag[MFRAG];
#pragma unroll
    for (int mi = 0; mi < MFRAG; ++mi) {
      const __bf16* pr = As + (wr * WM + mi * 16 + lrow) * LDK;
      v8bf lo = *(const v8bf*)(pr + lhi * 8);
      v8bf hi = *(const v8bf*)(pr + 16 + lhi * 8);
      afrag[mi] = __builtin_shufflevector(lo, hi, 0, 1, 2, 3, 4, 5, 6, 7,
                                          8, 9, 10, 11, 12, 13, 14, 15);
    }
    // B fragment (32x16 bf16): lane L holds col N=L%16; K 0..7,16..23 (L<16)
    // or K 8..15,24..31 (L>=16) -> two contiguous b128 reads.
    v16bf bfrag[NFRAG];
#pragma unroll
    for (int ni = 0; ni < NFRAG; ++ni) {
      const __bf16* pc = Bs + (wc * WN + ni * 16 + lrow) * LDK + lhi * 8;
      v8bf lo = *(const v8bf*)(pc);
      v8bf hi = *(const v8bf*)(pc + 16);
      bfrag[ni] = __builtin_shufflevector(lo, hi, 0, 1, 2, 3, 4, 5, 6, 7,
                                          8, 9, 10, 11, 12, 13, 14, 15);
    }

#pragma unroll
    for (int mi = 0; mi < MFRAG; ++mi)
#pragma unroll
      for (int ni = 0; ni < NFRAG; ++ni)
        acc[mi][ni] = __builtin_amdgcn_wmma_f32_16x16x32_bf16(
            false, afrag[mi], false, bfrag[ni], (short)0, acc[mi][ni],
            false, false);

    if (kt + 1 < nk) {             // fill the other buffer while it's idle
      stashA(nxt);
      stashB(nxt);
    }
    __syncthreads();
  }

  // Epilogue: C/D layout -> VGPR r holds row (r + 8*(lane>=16)), col = lane%16.
#pragma unroll
  for (int mi = 0; mi < MFRAG; ++mi) {
#pragma unroll
    for (int r = 0; r < 8; ++r) {
      const int row = bm0 + wr * WM + mi * 16 + r + 8 * lhi;
      const float s = (EPI == EPI_ROWSCALE) ? scale[row] : 1.0f;
#pragma unroll
      for (int ni = 0; ni < NFRAG; ++ni) {
        const int col = bn0 + wc * WN + ni * 16 + lrow;
        float v = acc[mi][ni][r];
        if (EPI == EPI_ROWSCALE) v *= s;
        if (EPI == EPI_RELU) v = fmaxf(v, 0.0f);
        C[(size_t)row * N + col] = v;
      }
    }
  }
}

// log_softmax over 64 columns; one wave per row (2 elems/lane + shfl_xor).
__global__ __launch_bounds__(256) void gwnn_logsoftmax64(
    const float* __restrict__ X, float* __restrict__ Y) {
  const int row  = blockIdx.x * 8 + (threadIdx.x >> 5);
  const int lane = threadIdx.x & 31;
  const float* x = X + (size_t)row * 64;
  float a = x[lane];
  float b = x[lane + 32];
  float m = fmaxf(a, b);
#pragma unroll
  for (int off = 16; off > 0; off >>= 1) m = fmaxf(m, __shfl_xor(m, off, 32));
  float s = __expf(a - m) + __expf(b - m);
#pragma unroll
  for (int off = 16; off > 0; off >>= 1) s += __shfl_xor(s, off, 32);
  const float lse = m + __logf(s);
  Y[(size_t)row * 64 + lane]      = a - lse;
  Y[(size_t)row * 64 + lane + 32] = b - lse;
}

extern "C" void kernel_launch(void* const* d_in, const int* in_sizes, int n_in,
                              void* d_out, int out_size, void* d_ws, size_t ws_size,
                              hipStream_t stream) {
  (void)in_sizes; (void)n_in; (void)out_size; (void)ws_size;
  const float* x    = (const float*)d_in[0];
  const float* wav  = (const float*)d_in[1];
  const float* wavi = (const float*)d_in[2];
  const float* w1   = (const float*)d_in[3];
  const float* f1   = (const float*)d_in[4];
  const float* w2   = (const float*)d_in[5];
  const float* f2   = (const float*)d_in[6];
  float* out = (float*)d_out;
  float* ws  = (float*)d_ws;

  constexpr int N = 8192, FIN = 512, H = 256, O = 64;
  float* bufA = ws;                      // [N,H]
  float* bufB = ws + (size_t)N * H;      // [N,H]
  float* bufC = ws + (size_t)2 * N * H;  // [N,H] (only [N,O] used)

  const dim3 blk(256);
  const dim3 gridW(1, N / 64);   // wide GEMMs: BN covers full output width
  const dim3 gridN(1, N / 64);   // narrow GEMMs (d = 64)

  // Layer 1: T1 = x @ w1
  gwnn_gemm<64, 256, 32, 64, EPI_NONE>
      <<<gridW, blk, 0, stream>>>(x, w1, bufA, nullptr, N, H, FIN);
  // S1 = diag(f1) @ (Winv @ T1)
  gwnn_gemm<64, 256, 32, 64, EPI_ROWSCALE>
      <<<gridW, blk, 0, stream>>>(wavi, bufA, bufB, f1, N, H, N);
  // Hm = relu(W @ S1)
  gwnn_gemm<64, 256, 32, 64, EPI_RELU>
      <<<gridW, blk, 0, stream>>>(wav, bufB, bufA, nullptr, N, H, N);

  // Layer 2: T2 = Hm @ w2
  gwnn_gemm<64, 64, 16, 32, EPI_NONE>
      <<<gridN, blk, 0, stream>>>(bufA, w2, bufC, nullptr, N, O, H);
  // U2 = diag(f2) @ (Winv @ T2)
  gwnn_gemm<64, 64, 16, 32, EPI_ROWSCALE>
      <<<gridN, blk, 0, stream>>>(wavi, bufC, bufB, f2, N, O, N);
  // logits = W @ U2
  gwnn_gemm<64, 64, 16, 32, EPI_NONE>
      <<<gridN, blk, 0, stream>>>(wav, bufB, bufC, nullptr, N, O, N);

  // log_softmax along axis=1 (64 columns)
  gwnn_logsoftmax64<<<N / 8, blk, 0, stream>>>(bufC, out);
}
